// ReverseRNNLayer_59665685676323
// MI455X (gfx1250) — matrix operand order repaired and checked by reading, not compiled, not run on hardware
//
#include <hip/hip_runtime.h>

#define HDIM 512
#define NB 32
#define NGATE 2048

typedef __attribute__((ext_vector_type(16))) __bf16 v16bf;
typedef __attribute__((ext_vector_type(8)))  float  v8f;

struct Bits256 { uint4 lo, hi; };

__device__ __forceinline__ unsigned short f2bf(float f) {
  unsigned int u = __builtin_bit_cast(unsigned int, f);
  u += 0x7FFFu + ((u >> 16) & 1u);       // round-to-nearest-even
  return (unsigned short)(u >> 16);
}

__device__ __forceinline__ v16bf make_v16bf(uint4 lo, uint4 hi) {
  Bits256 t{lo, hi};
  return __builtin_bit_cast(v16bf, t);
}

// A fragment: 16x32 (MxK), row-major source with leading dim ld.
__device__ __forceinline__ v16bf load_a(const unsigned short* base, int ld, int row, int k0, int lane) {
  int half = lane >> 4;
  const unsigned short* p = base + (size_t)row * ld + k0 + half * 8;
  uint4 lo = *(const uint4*)p;            // 8 bf16
  uint4 hi = *(const uint4*)(p + 16);     // 8 bf16, +16 elements
  return make_v16bf(lo, hi);
}

// B fragment: 32x16 (KxN) where B[k][n] = W[col0+n][k], W row-major with ld.
__device__ __forceinline__ v16bf load_b(const unsigned short* base, int ld, int col0, int k0, int lane) {
  int n  = lane & 15;
  int kb = k0 + ((lane >> 4) << 3);
  const unsigned short* p = base + (size_t)(col0 + n) * ld + kb;
  uint4 lo = *(const uint4*)p;
  uint4 hi = *(const uint4*)(p + 16);
  return make_v16bf(lo, hi);
}

// CDNA5 async DMA: global -> LDS, 16 bytes per active lane, tracked by ASYNCcnt.
__device__ __forceinline__ void async_copy_b128(unsigned lds_addr, const void* gaddr) {
  asm volatile("global_load_async_to_lds_b128 %0, %1, off"
               :: "v"(lds_addr), "v"(gaddr) : "memory");
}
__device__ __forceinline__ void wait_asynccnt0() {
  asm volatile("s_wait_asynccnt 0" ::: "memory");
}

__device__ __forceinline__ float sigf(float x)     { return 1.0f / (1.0f + __expf(-x)); }
__device__ __forceinline__ float tanhfast(float x) { return 2.0f / (1.0f + __expf(-2.0f * x)) - 1.0f; }

// ---------------- fp32 -> bf16 conversion ----------------
__global__ void cvt_kernel(const float* __restrict__ src, unsigned short* __restrict__ dst, size_t n) {
  size_t i = (size_t)blockIdx.x * blockDim.x + threadIdx.x;
  size_t stride = (size_t)gridDim.x * blockDim.x;
  for (; i < n; i += stride) dst[i] = f2bf(src[i]);
}

// ---------------- init: bs/off from mask, fused bias, state ----------------
__global__ void init_kernel(const unsigned char* __restrict__ mask,
                            const float* __restrict__ b_ih, const float* __restrict__ b_hh,
                            const float* __restrict__ h0, const float* __restrict__ c0,
                            int* bs, int* off, float* biasSum,
                            unsigned short* hbuf0, unsigned short* hbuf1,
                            float* h_f32, float* c_f32, int* bar, int T) {
  int tid = threadIdx.x;
  for (int t = tid; t < T; t += blockDim.x) {
    int s = 0;
    for (int b = 0; b < NB; ++b) s += mask[t * NB + b] ? 1 : 0;
    bs[t] = s;
  }
  __syncthreads();
  if (tid == 0) {
    int o = 0;
    for (int t = 0; t < T; ++t) { off[t] = o; o += bs[t]; }
    bar[0] = 0; bar[1] = 0;
  }
  for (int i = tid; i < NGATE; i += blockDim.x) biasSum[i] = b_ih[i] + b_hh[i];
  for (int i = tid; i < NB * HDIM; i += blockDim.x) {
    unsigned short hb = f2bf(h0[i]);
    hbuf0[i] = hb; hbuf1[i] = hb;
    h_f32[i] = h0[i];
    c_f32[i] = c0[i];
  }
}

// ---------------- phase 1: xg = packed @ w_ih^T + (b_ih+b_hh) ----------------
// A panel (16 x 512 bf16) staged into LDS once per block via async DMA,
// shared by all 4 waves; B streamed from global (w_ih stays hot in L2).
__global__ void __launch_bounds__(128) gemm_xg_kernel(
    const unsigned short* __restrict__ xb, const unsigned short* __restrict__ wihb,
    const float* __restrict__ biasSum, float* __restrict__ xg, int total) {
  __shared__ __align__(16) unsigned short alds[16 * HDIM];   // 16 KB
  int lane = threadIdx.x & 31;
  int wave = threadIdx.x >> 5;
  int m0 = blockIdx.x * 16;
  int n0 = blockIdx.y * 64 + wave * 16;

  // async-stage A panel: 1024 x 16B chunks, 8 per thread
  unsigned albase = (unsigned)(uintptr_t)alds;
  for (int c = threadIdx.x; c < (16 * HDIM * 2) / 16; c += 128) {
    int row = c >> 6;                       // 64 x 16B chunks per 1KB row
    int rb  = c & 63;
    int srow = m0 + row;
    if (srow >= total) srow = total - 1;    // clamp (guarded at store)
    const char* g = (const char*)(xb + (size_t)srow * HDIM) + rb * 16;
    async_copy_b128(albase + c * 16, g);
  }
  wait_asynccnt0();
  __syncthreads();

  v8f acc = {};
  for (int k = 0; k < HDIM; k += 32) {
    v16bf a = load_a(alds, HDIM, lane & 15, k, lane);     // ds_load_b128
    v16bf b = load_b(wihb, HDIM, n0, k, lane);            // global_load_b128
    acc = __builtin_amdgcn_wmma_f32_16x16x32_bf16(false, a, false, b, (short)0, acc, false, false);
  }
  int col = n0 + (lane & 15);
  float bias = biasSum[col];
  int mbase = m0 + ((lane < 16) ? 0 : 8);
#pragma unroll
  for (int j = 0; j < 8; ++j) {
    int m = mbase + j;
    if (m < total) xg[(size_t)m * NGATE + col] = acc[j] + bias;
  }
}

// ---------------- phase 2: persistent reverse recurrence ----------------
// 32 blocks (one per 16-wide hidden chunk) x 128 threads (4 waves = 4 gates).
// Per step: async-DMA the 32x512 bf16 h panel into LDS once (instead of 4x
// redundant global reads), WMMA from LDS, fused cell update, prefetch next
// step's x-gates, grid barrier.
__global__ void __launch_bounds__(128) lstm_reverse_kernel(
    const float* __restrict__ xg, const unsigned short* __restrict__ whhb,
    unsigned short* hbuf0, unsigned short* hbuf1,
    float* h_f32, float* c_f32, const float* __restrict__ h0,
    const int* __restrict__ bs, const int* __restrict__ off,
    float* __restrict__ out_packed, int T, int nblocks, int* bar) {
  __shared__ __align__(16) unsigned short hlds[NB * HDIM];  // 32 KB
  __shared__ float gates[4][NB][16];                        //  8 KB
  int lane = threadIdx.x & 31;
  int g = threadIdx.x >> 5;                    // wave index == gate index
  int n0 = blockIdx.x * 16;                    // hidden chunk
  int bcol0 = g * HDIM + n0;                   // w_hh row base for this wave
  unsigned hlbase = (unsigned)(uintptr_t)hlds;

  for (int s = 0; s < T; ++s) {
    int t = T - 1 - s;
    const unsigned short* hread = (s & 1) ? hbuf1 : hbuf0;
    unsigned short* hwrite      = (s & 1) ? hbuf0 : hbuf1;
    int bs_t = bs[t];
    int off_t = off[t];

    // async-stage h panel: 2048 x 16B chunks, 16 per thread
    for (int c = threadIdx.x; c < (NB * HDIM * 2) / 16; c += 128) {
      async_copy_b128(hlbase + c * 16, (const char*)hread + c * 16);
    }
    wait_asynccnt0();
    __syncthreads();

    v8f acc0 = {};
    v8f acc1 = {};
    for (int k = 0; k < HDIM; k += 32) {
      v16bf b  = load_b(whhb, HDIM, bcol0, k, lane);            // global (L2-hot)
      v16bf a0 = load_a(hlds, HDIM, (lane & 15), k, lane);      // ds_load_b128
      v16bf a1 = load_a(hlds, HDIM, 16 + (lane & 15), k, lane); // ds_load_b128
      acc0 = __builtin_amdgcn_wmma_f32_16x16x32_bf16(false, a0, false, b, (short)0, acc0, false, false);
      acc1 = __builtin_amdgcn_wmma_f32_16x16x32_bf16(false, a1, false, b, (short)0, acc1, false, false);
    }
    {
      int n = lane & 15;
      int mb = (lane < 16) ? 0 : 8;
#pragma unroll
      for (int j = 0; j < 8; ++j) {
        gates[g][mb + j][n]      = acc0[j];
        gates[g][16 + mb + j][n] = acc1[j];
      }
    }
    __syncthreads();

    // fused elementwise LSTM cell update for this block's 32x16 slice
    for (int e = threadIdx.x; e < NB * 16; e += 128) {
      int b  = e >> 4;
      int nn = e & 15;
      int nH = n0 + nn;
      int hcidx = b * HDIM + nH;
      if (b < bs_t) {
        size_t row = (size_t)(off_t + b);
        const float* xr = xg + row * NGATE + nH;
        float gi = xr[0]        + gates[0][b][nn];
        float gf = xr[HDIM]     + gates[1][b][nn];
        float gg = xr[2 * HDIM] + gates[2][b][nn];
        float go = xr[3 * HDIM] + gates[3][b][nn];
        float cn = sigf(gf) * c_f32[hcidx] + sigf(gi) * tanhfast(gg);
        float hn = sigf(go) * tanhfast(cn);
        c_f32[hcidx] = cn;
        h_f32[hcidx] = hn;
        out_packed[row * HDIM + nH] = hn;
        hwrite[hcidx] = f2bf(hn);
      } else {
        hwrite[hcidx] = f2bf(h0[hcidx]);   // inactive carry re-arms to h0
      }
    }

    // prefetch next step's x-gate rows while we sit in the barrier
    if (t > 0) {
      int bs_n = bs[t - 1];
      int off_n = off[t - 1];
      int b = threadIdx.x & 31;
      int gg2 = threadIdx.x >> 5;
      if (b < bs_n)
        __builtin_prefetch(xg + (size_t)(off_n + b) * NGATE + gg2 * HDIM + n0, 0, 1);
    }

    // grid-wide sense-reversing barrier (32 co-resident blocks)
    __syncthreads();
    if (threadIdx.x == 0) {
      __threadfence();
      int gen = __hip_atomic_load(bar + 1, __ATOMIC_RELAXED, __HIP_MEMORY_SCOPE_AGENT);
      int arrived = __hip_atomic_fetch_add(bar, 1, __ATOMIC_ACQ_REL, __HIP_MEMORY_SCOPE_AGENT);
      if (arrived == nblocks - 1) {
        __hip_atomic_store(bar, 0, __ATOMIC_RELAXED, __HIP_MEMORY_SCOPE_AGENT);
        __hip_atomic_store(bar + 1, gen + 1, __ATOMIC_RELEASE, __HIP_MEMORY_SCOPE_AGENT);
      } else {
        while (__hip_atomic_load(bar + 1, __ATOMIC_ACQUIRE, __HIP_MEMORY_SCOPE_AGENT) == gen) {
          __builtin_amdgcn_s_sleep(2);
        }
      }
    }
    __syncthreads();
  }
}

// ---------------- final: hT, cT into output tail ----------------
__global__ void final_kernel(const float* __restrict__ h_f32, const float* __restrict__ c_f32,
                             float* __restrict__ out, size_t packedElems) {
  int i = blockIdx.x * blockDim.x + threadIdx.x;
  if (i < NB * HDIM) {
    out[packedElems + i]             = h_f32[i];
    out[packedElems + NB * HDIM + i] = c_f32[i];
  }
}

extern "C" void kernel_launch(void* const* d_in, const int* in_sizes, int n_in,
                              void* d_out, int out_size, void* d_ws, size_t ws_size,
                              hipStream_t stream) {
  const float* packed = (const float*)d_in[0];
  const float* h0     = (const float*)d_in[1];
  const float* c0     = (const float*)d_in[2];
  const float* w_ih   = (const float*)d_in[3];
  const float* w_hh   = (const float*)d_in[4];
  const float* b_ih   = (const float*)d_in[5];
  const float* b_hh   = (const float*)d_in[6];
  const unsigned char* mask = (const unsigned char*)d_in[9];

  const int total = in_sizes[0] / HDIM;      // 38800
  const int T     = in_sizes[9] / NB;        // 1600

  // workspace carve-out (~365 MB; dominated by precomputed x-gates)
  char* ws = (char*)d_ws;
  size_t o = 0;
  auto take = [&](size_t bytes) -> char* {
    char* p = ws + o;
    o += (bytes + 255) & ~(size_t)255;
    return p;
  };
  float*          xg     = (float*)take((size_t)total * NGATE * sizeof(float));
  unsigned short* xb     = (unsigned short*)take((size_t)total * HDIM * 2);
  unsigned short* wihb   = (unsigned short*)take((size_t)NGATE * HDIM * 2);
  unsigned short* whhb   = (unsigned short*)take((size_t)NGATE * HDIM * 2);
  unsigned short* hbuf0  = (unsigned short*)take((size_t)NB * HDIM * 2);
  unsigned short* hbuf1  = (unsigned short*)take((size_t)NB * HDIM * 2);
  float*          h_f32  = (float*)take((size_t)NB * HDIM * sizeof(float));
  float*          c_f32  = (float*)take((size_t)NB * HDIM * sizeof(float));
  float*          biasSum= (float*)take((size_t)NGATE * sizeof(float));
  int*            bs     = (int*)take((size_t)T * sizeof(int));
  int*            off    = (int*)take((size_t)T * sizeof(int));
  int*            bar    = (int*)take(256);
  (void)ws_size; (void)n_in; (void)out_size;

  init_kernel<<<1, 256, 0, stream>>>(mask, b_ih, b_hh, h0, c0, bs, off, biasSum,
                                     hbuf0, hbuf1, h_f32, c_f32, bar, T);
  cvt_kernel<<<2048, 256, 0, stream>>>(packed, xb, (size_t)total * HDIM);
  cvt_kernel<<<512, 256, 0, stream>>>(w_ih, wihb, (size_t)NGATE * HDIM);
  cvt_kernel<<<512, 256, 0, stream>>>(w_hh, whhb, (size_t)NGATE * HDIM);

  dim3 g1((total + 15) / 16, NGATE / 64);
  gemm_xg_kernel<<<g1, 128, 0, stream>>>(xb, wihb, biasSum, xg, total);

  lstm_reverse_kernel<<<HDIM / 16, 128, 0, stream>>>(xg, whhb, hbuf0, hbuf1,
                                                     h_f32, c_f32, h0, bs, off,
                                                     (float*)d_out, T, HDIM / 16, bar);

  final_kernel<<<(NB * HDIM + 255) / 256, 256, 0, stream>>>(h_f32, c_f32, (float*)d_out,
                                                            (size_t)total * HDIM);
}
